// ScaledDotProductAttention_61383672594486
// MI455X (gfx1250) — matrix operand order repaired and checked by reading, not compiled, not run on hardware
//
#include <hip/hip_runtime.h>
#include <math.h>

// ScaledDotProductAttention for MI455X (gfx1250, wave32, WMMA).
// B=2, L=2048, N=16 heads, D=64. fp32 I/O, f16 WMMA with f32 accumulate.
//
// Grid: one block per (batch, 16-row query tile) = 2*128 = 256 blocks.
// Block: 512 threads = 16 waves; wave w owns head n = w.
// Two-pass softmax: pass 1 keeps PER-LANE online (max, sum) in registers
// (no cross-lane traffic in the hot loop), merged once with a 4-step xor
// butterfly (exact associative combine). Pass 2 recomputes QK^T, stages P
// through LDS [i][j][n] for coalesced stores of the bandwidth-dominant
// 512MB attention tensor, and accumulates context with PV WMMAs.

typedef __attribute__((ext_vector_type(16))) _Float16 v16h;
typedef __attribute__((ext_vector_type(8)))  float    v8f;

#define ATTN_L  2048
#define ATTN_NH 16
#define ATTN_D  64
#define ATTN_SCALE 0.125f
#define ATTN_NEGINF (-1.0e9f)

// One 16x16 score tile S = Q_tile * K_tile^T for head n.
// aq0/aq1: Q A-fragments (16x32 f16, d=0..31 / 32..63).
__device__ __forceinline__ v8f qk_tile(const v16h aq0, const v16h aq1,
                                       const float* __restrict__ Kb,
                                       int j0, int l16, int h, int n) {
  // B-matrix 32x16 (f16): lane holds col j = l16, rows k = 8h + t + (t & 8).
  const float* krow = Kb + (((size_t)(j0 + l16)) * ATTN_NH + n) * ATTN_D;
  v16h bk0, bk1;
#pragma unroll
  for (int t = 0; t < 16; ++t) {
    bk0[t] = (_Float16)krow[8 * h + t + (t & 8)];       // d = 8h+t+(t&8)
    bk1[t] = (_Float16)krow[32 + 8 * h + t + (t & 8)];  // d = 32+8h+t+(t&8)
  }
  v8f c = {};
  c = __builtin_amdgcn_wmma_f32_16x16x32_f16(false, aq0, false, bk0,
                                             (short)0, c, false, false);
  c = __builtin_amdgcn_wmma_f32_16x16x32_f16(false, aq1, false, bk1,
                                             (short)0, c, false, false);
  return c;
}

__global__ __launch_bounds__(512) void
ScaledDotProductAttention_61383672594486_kernel(
    const float* __restrict__ Q, const float* __restrict__ K,
    const float* __restrict__ V, const unsigned char* __restrict__ Mask,
    float* __restrict__ Octx, float* __restrict__ Oattn) {
  constexpr int L = ATTN_L, NH = ATTN_NH, D = ATTN_D;

  // P staging tile: [i=16][j=32][n=16] f32 = 32 KB (of 320 KB/WGP).
  __shared__ float lds_p[16 * 32 * 16];

  const int tid  = threadIdx.x;
  const int lane = tid & 31;
  const int l16  = lane & 15;   // column index within fragments
  const int h    = lane >> 4;   // half-wave select
  const int n    = tid >> 5;    // head = wave id (0..15)

  const int b  = blockIdx.x >> 7;           // 128 i-tiles per batch
  const int i0 = (blockIdx.x & 127) << 4;   // query-tile start row

  const float* Qb = Q + (size_t)b * L * NH * D;
  const float* Kb = K + (size_t)b * L * NH * D;
  const float* Vb = V + (size_t)b * L * NH * D;
  const unsigned char* Mb = Mask + (size_t)b * L * L;

  // ---- Q A-fragments, resident for the whole kernel.
  // A 16x32 f16 layout: lane row m = l16; halfs t<8 -> k = 8h+t,
  // halfs t>=8 -> k = 8h+16+(t-8). Second fragment: +32.
  const float* qrow = Qb + (((size_t)(i0 + l16)) * NH + n) * D;
  v16h aq0, aq1;
#pragma unroll
  for (int t = 0; t < 8; ++t) {
    aq0[t]     = (_Float16)qrow[8 * h + t];
    aq0[t + 8] = (_Float16)qrow[8 * h + 16 + t];
    aq1[t]     = (_Float16)qrow[8 * h + 32 + t];
    aq1[t + 8] = (_Float16)qrow[8 * h + 48 + t];
  }

  // ---- Pass 1: PER-LANE online softmax stats, registers only.
  // C/D layout: lane col j = l16, rows m = r + 8h for VGPR r=0..7.
  float runmax[8], runsum[8];
#pragma unroll
  for (int r = 0; r < 8; ++r) { runmax[r] = -INFINITY; runsum[r] = 0.0f; }

  for (int jc = 0; jc < L / 16; ++jc) {
    const int j0 = jc * 16;
    if (jc + 1 < L / 16) {  // prefetch next key tile row (global_prefetch_b8)
      __builtin_prefetch(Kb + (((size_t)(j0 + 16 + l16)) * NH + n) * D, 0, 0);
    }
    v8f c = qk_tile(aq0, aq1, Kb, j0, l16, h, n);
#pragma unroll
    for (int r = 0; r < 8; ++r) {
      const int mrow = r + 8 * h;
      float s = c[r] * ATTN_SCALE;
      const unsigned char mv = Mb[(size_t)(i0 + mrow) * L + j0 + l16];
      s = mv ? ATTN_NEGINF : s;
      const float nm = fmaxf(runmax[r], s);
      runsum[r] = runsum[r] * __expf(runmax[r] - nm) + __expf(s - nm);
      runmax[r] = nm;
    }
  }

  // ---- Single cross-lane merge per row (xor butterfly, exact combine).
  float rinv[8];
#pragma unroll
  for (int r = 0; r < 8; ++r) {
    float m = runmax[r], s = runsum[r];
#pragma unroll
    for (int msk = 1; msk <= 8; msk <<= 1) {
      const float m2 = __shfl_xor(m, msk);
      const float s2 = __shfl_xor(s, msk);
      const float nm = fmaxf(m, m2);
      s = s * __expf(m - nm) + s2 * __expf(m2 - nm);
      m = nm;
    }
    runmax[r] = m;
    rinv[r]   = 1.0f / s;
  }

  // ---- Pass 2: recompute S, emit P, accumulate O = P*V.
  v8f o[4];
#pragma unroll
  for (int dd = 0; dd < 4; ++dd) o[dd] = (v8f){};

  for (int jb = 0; jb < L / 32; ++jb) {
#pragma unroll
    for (int sub = 0; sub < 2; ++sub) {
      const int j0 = jb * 32 + sub * 16;
      v8f c = qk_tile(aq0, aq1, Kb, j0, l16, h, n);
#pragma unroll
      for (int r = 0; r < 8; ++r) {
        const int mrow = r + 8 * h;
        float s = c[r] * ATTN_SCALE;
        const unsigned char mv = Mb[(size_t)(i0 + mrow) * L + j0 + l16];
        s = mv ? ATTN_NEGINF : s;
        const float p = __expf(s - runmax[r]) * rinv[r];
        lds_p[(mrow * 32 + (sub * 16 + l16)) * 16 + n] = p;
      }
    }
    __syncthreads();

    // Coalesced store of the attention tile: for row m, [j][n] is a
    // contiguous 512-float segment both in LDS and in global memory.
    {
      const int m = tid >> 5;
      const float4* src = (const float4*)(lds_p + m * 512);
      float4* dst = (float4*)(Oattn +
          (((size_t)(b * L + i0 + m)) * L + (size_t)jb * 32) * NH);
#pragma unroll
      for (int k = 0; k < 4; ++k) dst[lane + 32 * k] = src[lane + 32 * k];
    }

    // P A-fragment (16x32, K = the 32 key positions of this block).
    v16h ap;
#pragma unroll
    for (int t = 0; t < 8; ++t) {
      ap[t]     = (_Float16)lds_p[(l16 * 32 + (8 * h + t)) * 16 + n];
      ap[t + 8] = (_Float16)lds_p[(l16 * 32 + (8 * h + 16 + t)) * 16 + n];
    }

    // V B-fragments (32x16 per d-block): lane col d = dd*16 + l16,
    // rows k = 8h + t + (t & 8) -> key j = jb*32 + 8h + t + (t & 8).
#pragma unroll
    for (int dd = 0; dd < 4; ++dd) {
      v16h bv;
#pragma unroll
      for (int t = 0; t < 16; ++t) {
        bv[t] = (_Float16)Vb[(((size_t)(jb * 32 + 8 * h + t + (t & 8))) * NH + n) * D +
                             dd * 16 + l16];
      }
      o[dd] = __builtin_amdgcn_wmma_f32_16x16x32_f16(false, ap, false, bv,
                                                     (short)0, o[dd], false,
                                                     false);
    }
    __syncthreads();  // protect lds_p before next iteration overwrites it
  }

  // ---- Context output: O[m, d], lane col d = dd*16 + l16, row m = r + 8h.
#pragma unroll
  for (int dd = 0; dd < 4; ++dd) {
#pragma unroll
    for (int r = 0; r < 8; ++r) {
      const int mrow = r + 8 * h;
      Octx[(((size_t)(b * L + i0 + mrow)) * NH + n) * D + dd * 16 + l16] =
          o[dd][r];
    }
  }
}

extern "C" void kernel_launch(void* const* d_in, const int* in_sizes, int n_in,
                              void* d_out, int out_size, void* d_ws,
                              size_t ws_size, hipStream_t stream) {
  (void)in_sizes; (void)n_in; (void)out_size; (void)d_ws; (void)ws_size;
  const float* Q = (const float*)d_in[0];
  const float* K = (const float*)d_in[1];
  const float* V = (const float*)d_in[2];
  // jnp.bool_ mask -> 1 byte per element
  const unsigned char* Mask = (const unsigned char*)d_in[3];

  float* ctx  = (float*)d_out;                                   // [B,L,N,D]
  float* attn = ctx + (size_t)2 * ATTN_L * ATTN_NH * ATTN_D;     // [B,L,L,N]

  const int blocks = 2 * (ATTN_L / 16);  // 256
  ScaledDotProductAttention_61383672594486_kernel<<<dim3(blocks), dim3(512), 0,
                                                    stream>>>(Q, K, V, Mask,
                                                              ctx, attn);
}
